// MultiHeadSelfAttention_20418274525415
// MI455X (gfx1250) — compile-verified, matrix operand order checked by reading, not run on hardware
//
#include <hip/hip_runtime.h>
#include <hip/hip_bf16.h>

// ---------------------------------------------------------------------------
// MHA forward for MI455X (gfx1250): bf16 WMMA + Tensor Data Mover staging
// Stage 1: f32 -> bf16 conversion of x, Wqkv, Wout
// Stage 2: qkv = x @ Wqkv + bqkv    (bf16 in, bf16 out, f32 accum)
// Stage 3: flash attention per (b,h), 128 q-rows per block, bf16 ctx out
// Stage 4: out = ctx @ Wout + bout  (bf16 in, f32 out)
// ---------------------------------------------------------------------------

typedef __bf16 bf16_t;
typedef __attribute__((ext_vector_type(16))) __bf16 v16bf;
typedef __attribute__((ext_vector_type(8)))  __bf16 bf16x8;
typedef __attribute__((ext_vector_type(8)))  float  v8f;

union V16U { v16bf v; bf16x8 h[2]; };

#define WMMA_BF16(a, b, c) \
    __builtin_amdgcn_wmma_f32_16x16x32_bf16(false, (a), false, (b), (short)0, (c), false, false)

static constexpr int B_  = 16;
static constexpr int T_  = 512;
static constexpr int H_  = 1024;
static constexpr int NH_ = 16;
static constexpr int HD_ = 64;
static constexpr float SCALE_ = 0.125f; // 1/sqrt(64)

// ---------------------------------------------------------------------------
// Tensor Data Mover: 2D tile Global -> LDS with LDS row padding.
// D# layout per CDNA5 ISA ch.8 (group0 128b, group1 256b; groups 2/3 zero for 2D).
// ---------------------------------------------------------------------------
#if defined(__has_builtin)
#if __has_builtin(__builtin_amdgcn_tensor_load_to_lds)
#define HAVE_TDM 1
#endif
#endif

#ifdef HAVE_TDM
typedef __attribute__((ext_vector_type(4))) unsigned int u32x4;
typedef __attribute__((ext_vector_type(8))) int          i32x8;
typedef __attribute__((ext_vector_type(4))) int          i32x4;

// pad_interval_code: pad after 2^(c+1) DWORDs; pad_amount_code: pad (c+1) DWORDs
__device__ __forceinline__ void tdm_load_2d(
    unsigned lds_byte_addr, const void* gaddr,
    unsigned tensor_d0, unsigned tensor_d1, unsigned long long d0_stride,
    unsigned tile_d0, unsigned tile_d1,
    unsigned pad_interval_code, unsigned pad_amount_code)
{
    unsigned long long ga = (unsigned long long)(size_t)gaddr;
    u32x4 g0;
    g0.x = 1u;                                              // count=1 (valid user D#)
    g0.y = lds_byte_addr;                                   // [63:32] lds_addr
    g0.z = (unsigned)(ga & 0xffffffffu);                    // [95:64] global_addr lo
    g0.w = (unsigned)((ga >> 32) & 0x01ffffffu)             // [120:96] global_addr hi
         | (2u << 30);                                      // [127:126] type=2 (image)
    i32x8 g1;
    g1[0] = (int)((1u << 16)                                // data_size=1 -> 2 bytes
                | (1u << 20)                                // pad_enable
                | (pad_interval_code << 22)
                | (pad_amount_code << 25));
    g1[1] = (int)((tensor_d0 & 0xffffu) << 16);             // tensor_dim0[15:0]
    g1[2] = (int)(((tensor_d0 >> 16) & 0xffffu) | ((tensor_d1 & 0xffffu) << 16));
    g1[3] = (int)(((tensor_d1 >> 16) & 0xffffu) | ((tile_d0 & 0xffffu) << 16));
    g1[4] = (int)(tile_d1 & 0xffffu);                       // tile_dim1; tile_dim2=0
    g1[5] = (int)(unsigned)(d0_stride & 0xffffffffu);       // dim0_stride lo32
    g1[6] = (int)(unsigned)((d0_stride >> 32) & 0xffffu);   // dim0_stride hi16
    g1[7] = 0;
    i32x4 z4 = {0, 0, 0, 0};
#if __clang_major__ >= 23
    i32x8 z8 = {0, 0, 0, 0, 0, 0, 0, 0};
    __builtin_amdgcn_tensor_load_to_lds(g0, g1, z4, z4, z8, 0);
#else
    __builtin_amdgcn_tensor_load_to_lds(g0, g1, z4, z4, 0);
#endif
}

// low 32 bits of a generic LDS pointer == LDS byte offset (ISA aperture rule)
__device__ __forceinline__ unsigned lds_off(const void* p) {
    return (unsigned)(size_t)p;
}
#endif // HAVE_TDM

// ---------------------------------------------------------------------------
// Stage 1: fp32 -> bf16 conversion
// ---------------------------------------------------------------------------
__global__ void cvt_f32_bf16(const float* __restrict__ in, bf16_t* __restrict__ out, int n) {
    int i = blockIdx.x * blockDim.x + threadIdx.x;
    int stride = gridDim.x * blockDim.x;
    for (; i < n; i += stride) out[i] = (bf16_t)in[i];
}

// ---------------------------------------------------------------------------
// Stage 2/4: C[M,N] = A[M,K] @ B[K,N] + bias.  A,B bf16; C f32 or bf16.
// Block: 256 threads (8 wave32s). Tile 128x128x32.
// A tile staged by TDM (pad feature gives the 40-elem LDS stride); B staged
// manually (transposed so B-fragments are contiguous b128 LDS reads).
// ---------------------------------------------------------------------------
template<bool OUT_BF16>
__global__ __launch_bounds__(256) void gemm_bf16_128x128(
    const bf16_t* __restrict__ A, const bf16_t* __restrict__ Bm,
    const float* __restrict__ bias, void* __restrict__ Cout,
    int M, int N, int K)
{
    __shared__ bf16_t As[128][40];  // row-major A tile, +8 pad
    __shared__ bf16_t Bs[128][40];  // B tile stored transposed: [n][k]

    const int tid  = threadIdx.x;
    const int lane = tid & 31;
    const int w    = tid >> 5;
    const int wm   = w & 3;
    const int wn   = w >> 2;
    const int hf   = (lane >> 4) & 1;   // lane half (K-split in WMMA layout)
    const int lc   = lane & 15;

    const int m_blk = blockIdx.y * 128;
    const int n_blk = blockIdx.x * 128;

    const int bk   = tid >> 3;          // 0..31
    const int bn   = (tid & 7) * 16;    // 0..112
#ifndef HAVE_TDM
    const int arow = tid >> 1;          // 0..127
    const int acol = (tid & 1) * 16;    // 0 or 16
#endif

    v8f acc[2][4];
    #pragma unroll
    for (int i = 0; i < 2; i++)
        #pragma unroll
        for (int j = 0; j < 4; j++)
            acc[i][j] = (v8f){0.f,0.f,0.f,0.f,0.f,0.f,0.f,0.f};

    for (int k0 = 0; k0 < K; k0 += 32) {
        // stage A tile: TDM DMA, 128 rows x 32 bf16, LDS rows padded 64B+16B
#ifdef HAVE_TDM
        if (w == 0) {
            tdm_load_2d(lds_off(&As[0][0]),
                        A + (size_t)m_blk * K + k0,
                        /*tensor_d0=*/32, /*tensor_d1=*/128, /*d0_stride=*/(unsigned long long)K,
                        /*tile_d0=*/32, /*tile_d1=*/128,
                        /*pad_interval(16 DW)=*/3, /*pad_amount(4 DW)=*/3);
        }
#else
        {
            const bf16_t* src = A + (size_t)(m_blk + arow) * K + k0 + acol;
            bf16x8 v0 = *(const bf16x8*)(src);
            bf16x8 v1 = *(const bf16x8*)(src + 8);
            *(bf16x8*)&As[arow][acol]     = v0;
            *(bf16x8*)&As[arow][acol + 8] = v1;
        }
#endif
        // stage B tile transposed (manual: TDM cannot transpose)
        {
            const bf16_t* src = Bm + (size_t)(k0 + bk) * N + n_blk + bn;
            bf16x8 v0 = *(const bf16x8*)(src);
            bf16x8 v1 = *(const bf16x8*)(src + 8);
            #pragma unroll
            for (int i = 0; i < 8; i++) Bs[bn + i][bk]     = v0[i];
            #pragma unroll
            for (int i = 0; i < 8; i++) Bs[bn + 8 + i][bk] = v1[i];
            if (k0 + 32 < K)  // prefetch next B tile into GL2
                __builtin_prefetch(Bm + (size_t)(k0 + 32 + bk) * N + n_blk + bn, 0, 0);
        }
#ifdef HAVE_TDM
        if (w == 0) __builtin_amdgcn_s_wait_tensorcnt((short)0);
#endif
        __syncthreads();

        // fragment loads per ISA 16-bit A/B layout:
        // lane -> row (A) / col (B) = lc; lane-half selects K sub-chunks {hf*8, 16+hf*8}
        V16U a[2], b[4];
        #pragma unroll
        for (int i = 0; i < 2; i++) {
            int r = wm * 32 + i * 16 + lc;
            a[i].h[0] = *(const bf16x8*)&As[r][hf * 8];
            a[i].h[1] = *(const bf16x8*)&As[r][16 + hf * 8];
        }
        #pragma unroll
        for (int j = 0; j < 4; j++) {
            int r = wn * 64 + j * 16 + lc;
            b[j].h[0] = *(const bf16x8*)&Bs[r][hf * 8];
            b[j].h[1] = *(const bf16x8*)&Bs[r][16 + hf * 8];
        }
        #pragma unroll
        for (int i = 0; i < 2; i++)
            #pragma unroll
            for (int j = 0; j < 4; j++)
                acc[i][j] = WMMA_BF16(a[i].v, b[j].v, acc[i][j]);

        __syncthreads();
    }

    // epilogue: C layout -> element (m,n): vgpr r = m%8, lane = (m/8)*16 + n
    #pragma unroll
    for (int i = 0; i < 2; i++) {
        #pragma unroll
        for (int j = 0; j < 4; j++) {
            #pragma unroll
            for (int r = 0; r < 8; r++) {
                int m = m_blk + wm * 32 + i * 16 + r + hf * 8;
                int n = n_blk + wn * 64 + j * 16 + lc;
                float v = acc[i][j][r] + bias[n];
                if (OUT_BF16) ((bf16_t*)Cout)[(size_t)m * N + n] = (bf16_t)v;
                else          ((float*) Cout)[(size_t)m * N + n] = v;
            }
        }
    }
}

// ---------------------------------------------------------------------------
// Stage 3: flash attention. Block = (b,h) x 128 q-rows; wave owns 16 q-rows.
// qkv layout per row t: [3][NH][HD] (Q/K/V at +0H/+1H/+2H, head at h*HD).
// K tile staged by TDM (padded rows 128B+16B -> 72-elem stride); V staged
// manually transposed for contiguous PV B-fragments.
// ---------------------------------------------------------------------------
__global__ __launch_bounds__(256) void flash_attn(
    const bf16_t* __restrict__ qkv, const int* __restrict__ mask,
    bf16_t* __restrict__ ctx)
{
    __shared__ bf16_t Klds[32][72];     // [key][hd], shared by all 8 waves
    __shared__ bf16_t Vt[64][40];       // V transposed: [hd][key]
    __shared__ bf16_t Ps[8][16][40];    // wave-private P re-layout scratch

    const int tid  = threadIdx.x;
    const int lane = tid & 31;
    const int w    = tid >> 5;
    const int hf   = (lane >> 4) & 1;
    const int lc   = lane & 15;

    const int bh = blockIdx.x;
    const int b  = bh / NH_;
    const int h  = bh % NH_;
    const int q0 = blockIdx.y * 128 + w * 16;

    const size_t row_stride = (size_t)3 * H_;
    const bf16_t* Qbase = qkv + (size_t)(b * T_) * row_stride + 0 * H_ + h * HD_;
    const bf16_t* Kbase = qkv + (size_t)(b * T_) * row_stride + 1 * H_ + h * HD_;
    const bf16_t* Vbase = qkv + (size_t)(b * T_) * row_stride + 2 * H_ + h * HD_;

    // Q fragments for both 32-wide hd chunks, resident for the whole kernel
    V16U qa[2];
    {
        const bf16_t* qrow = Qbase + (size_t)(q0 + lc) * row_stride;
        #pragma unroll
        for (int c = 0; c < 2; c++) {
            qa[c].h[0] = *(const bf16x8*)(qrow + c * 32 + hf * 8);
            qa[c].h[1] = *(const bf16x8*)(qrow + c * 32 + 16 + hf * 8);
        }
    }

    v8f o[4];
    #pragma unroll
    for (int j = 0; j < 4; j++) o[j] = (v8f){0.f,0.f,0.f,0.f,0.f,0.f,0.f,0.f};
    float mrow[8], lrow[8];
    #pragma unroll
    for (int r = 0; r < 8; r++) { mrow[r] = -1e30f; lrow[r] = 0.f; }

    const int key_ld = tid >> 3;        // 0..31
    const int col_ld = (tid & 7) * 8;   // 0..56

    for (int kb = 0; kb < T_; kb += 32) {
        // stage K chunk via Tensor Data Mover (32 keys x 64 bf16, strided rows)
#ifdef HAVE_TDM
        if (w == 0) {
            tdm_load_2d(lds_off(&Klds[0][0]),
                        Kbase + (size_t)kb * row_stride,
                        /*tensor_d0=*/64, /*tensor_d1=*/32, /*d0_stride=*/(unsigned long long)row_stride,
                        /*tile_d0=*/64, /*tile_d1=*/32,
                        /*pad_interval(32 DW)=*/4, /*pad_amount(4 DW)=*/3);
        }
#else
        {
            bf16x8 kv = *(const bf16x8*)(Kbase + (size_t)(kb + key_ld) * row_stride + col_ld);
            *(bf16x8*)&Klds[key_ld][col_ld] = kv;
        }
#endif
        // stage V chunk transposed (manual)
        {
            bf16x8 vv = *(const bf16x8*)(Vbase + (size_t)(kb + key_ld) * row_stride + col_ld);
            #pragma unroll
            for (int i = 0; i < 8; i++) Vt[col_ld + i][key_ld] = vv[i];
        }
#ifdef HAVE_TDM
        if (w == 0) __builtin_amdgcn_s_wait_tensorcnt((short)0);
#endif
        __syncthreads();

        // S = Q K^T  (two 16-key tiles, hd reduced in two 32-chunks)
        v8f s[2];
        #pragma unroll
        for (int t = 0; t < 2; t++) {
            V16U bk0, bk1;
            int krow = t * 16 + lc;
            bk0.h[0] = *(const bf16x8*)&Klds[krow][hf * 8];
            bk0.h[1] = *(const bf16x8*)&Klds[krow][16 + hf * 8];
            bk1.h[0] = *(const bf16x8*)&Klds[krow][32 + hf * 8];
            bk1.h[1] = *(const bf16x8*)&Klds[krow][48 + hf * 8];
            v8f z = (v8f){0.f,0.f,0.f,0.f,0.f,0.f,0.f,0.f};
            s[t] = WMMA_BF16(qa[0].v, bk0.v, z);
            s[t] = WMMA_BF16(qa[1].v, bk1.v, s[t]);
        }

        // scale + mask (mask column = key index = lane column)
        const int mv0 = mask[b * T_ + kb + lc];
        const int mv1 = mask[b * T_ + kb + 16 + lc];
        #pragma unroll
        for (int r = 0; r < 8; r++) {
            float x0 = s[0][r] * SCALE_; if (mv0 == 0) x0 = -1e30f;
            float x1 = s[1][r] * SCALE_; if (mv1 == 0) x1 = -1e30f;
            s[0][r] = x0; s[1][r] = x1;
        }

        // online softmax: row reductions across the 16 lanes of each half
        float alpha[8];
        #pragma unroll
        for (int r = 0; r < 8; r++) {
            float bm = fmaxf(s[0][r], s[1][r]);
            #pragma unroll
            for (int off = 1; off < 16; off <<= 1)
                bm = fmaxf(bm, __shfl_xor(bm, off, 16));
            float mnew = fmaxf(mrow[r], bm);
            alpha[r] = __expf(mrow[r] - mnew);
            mrow[r] = mnew;
            float p0 = __expf(s[0][r] - mnew);
            float p1 = __expf(s[1][r] - mnew);
            s[0][r] = p0; s[1][r] = p1;
            float rs = p0 + p1;
            #pragma unroll
            for (int off = 1; off < 16; off <<= 1)
                rs += __shfl_xor(rs, off, 16);
            lrow[r] = lrow[r] * alpha[r] + rs;
        }

        // rescale accumulators
        #pragma unroll
        for (int j = 0; j < 4; j++)
            #pragma unroll
            for (int r = 0; r < 8; r++)
                o[j][r] *= alpha[r];

        // re-layout P (C layout -> A-fragment layout) via wave-private LDS
        #pragma unroll
        for (int r = 0; r < 8; r++) {
            int m = r + hf * 8;
            Ps[w][m][lc]      = (bf16_t)s[0][r];
            Ps[w][m][16 + lc] = (bf16_t)s[1][r];
        }
        V16U pa;  // DS ops in-order within wave: safe without barrier
        pa.h[0] = *(const bf16x8*)&Ps[w][lc][hf * 8];
        pa.h[1] = *(const bf16x8*)&Ps[w][lc][16 + hf * 8];

        // O += P @ V  (V B-frags contiguous from transposed tile)
        #pragma unroll
        for (int j = 0; j < 4; j++) {
            V16U vb;
            int vr = j * 16 + lc;
            vb.h[0] = *(const bf16x8*)&Vt[vr][hf * 8];
            vb.h[1] = *(const bf16x8*)&Vt[vr][16 + hf * 8];
            o[j] = WMMA_BF16(pa.v, vb.v, o[j]);
        }

        __syncthreads();
    }

    // normalize and store ctx (bf16, [B,T,H] with head offset h*HD)
    #pragma unroll
    for (int r = 0; r < 8; r++) {
        float inv = 1.0f / lrow[r];
        int q = q0 + r + hf * 8;
        bf16_t* dst = ctx + (size_t)(b * T_ + q) * H_ + h * HD_;
        #pragma unroll
        for (int j = 0; j < 4; j++)
            dst[j * 16 + lc] = (bf16_t)(o[j][r] * inv);
    }
}

// ---------------------------------------------------------------------------
// host launcher
// ---------------------------------------------------------------------------
extern "C" void kernel_launch(void* const* d_in, const int* in_sizes, int n_in,
                              void* d_out, int out_size, void* d_ws, size_t ws_size,
                              hipStream_t stream) {
    (void)in_sizes; (void)n_in; (void)out_size; (void)ws_size;

    const float* x    = (const float*)d_in[0];
    const int*   mask = (const int*)  d_in[1];
    const float* Wqkv = (const float*)d_in[2];
    const float* bqkv = (const float*)d_in[3];
    const float* Wout = (const float*)d_in[4];
    const float* bout = (const float*)d_in[5];
    float* out = (float*)d_out;

    const int nX    = B_ * T_ * H_;        // 8,388,608
    const int nWqkv = H_ * 3 * H_;         // 3,145,728
    const int nWout = H_ * H_;             // 1,048,576

    char* w = (char*)d_ws;
    bf16_t* x_bf    = (bf16_t*)(w);
    bf16_t* wqkv_bf = (bf16_t*)(w + 16777216);
    bf16_t* wout_bf = (bf16_t*)(w + 16777216 + 6291456);
    bf16_t* qkv_bf  = (bf16_t*)(w + 16777216 + 6291456 + 2097152);
    bf16_t* ctx_bf  = (bf16_t*)(w + 16777216 + 6291456 + 2097152 + 50331648);

    cvt_f32_bf16<<<2048, 256, 0, stream>>>(x,    x_bf,    nX);
    cvt_f32_bf16<<<2048, 256, 0, stream>>>(Wqkv, wqkv_bf, nWqkv);
    cvt_f32_bf16<<<1024, 256, 0, stream>>>(Wout, wout_bf, nWout);

    // qkv = x @ Wqkv + bqkv   (M=8192, N=3072, K=1024) -> bf16
    gemm_bf16_128x128<true><<<dim3(3072 / 128, 8192 / 128), 256, 0, stream>>>(
        x_bf, wqkv_bf, bqkv, (void*)qkv_bf, 8192, 3072, 1024);

    // flash attention -> ctx bf16
    flash_attn<<<dim3(B_ * NH_, T_ / 128), 256, 0, stream>>>(qkv_bf, mask, ctx_bf);

    // out = ctx @ Wout + bout (M=8192, N=1024, K=1024) -> f32
    gemm_bf16_128x128<false><<<dim3(1024 / 128, 8192 / 128), 256, 0, stream>>>(
        ctx_bf, wout_bf, bout, (void*)out, 8192, 1024, 1024);
}
